// Luong_61684320305412
// MI455X (gfx1250) — matrix operand order repaired and checked by reading, not compiled, not run on hardware
//
#include <hip/hip_runtime.h>

// CDNA5 / gfx1250 flash-style dual-softmax Luong attention, round 3:
//  - K / K^T tiles staged block-wide in LDS via async global->LDS copies
//    (double-buffered, ASYNCcnt + workgroup barrier per iteration)
//  - padded LDS strides to kill the 16-way bank conflicts of round 1
//  - all WMMA operands sourced from LDS
//  - fix: __builtin_amdgcn_global_load_async_to_lds_b128 takes
//    (AS1 int4*, AS3 int4*, int offset, int cpol)

typedef __attribute__((ext_vector_type(16))) _Float16 v16h;
typedef __attribute__((ext_vector_type(8)))  _Float16 v8h;
typedef __attribute__((ext_vector_type(8)))  float    v8f;
typedef __attribute__((__vector_size__(16))) int      vint4;

#define HD        256
#define NEG_BIG   (-100000000.0f)
#define SCALE     0.0625f            // 1/sqrt(256)
#define WAVES     4                  // 128 threads / block, 16 q-rows per wave
#define QS_STRIDE 264                // 256 + 8 halfs pad -> 132 dwords == 4 mod 64
#define KS_STRIDE 264
#define KT_STRIDE 40                 // 32 + 8 halfs pad  -> 20 dwords

// ---- async 16B global -> LDS copy (ASYNCcnt tracked) -----------------------
__device__ __forceinline__ void async_cp16(const _Float16* g, _Float16* l) {
#if __has_builtin(__builtin_amdgcn_global_load_async_to_lds_b128)
    __builtin_amdgcn_global_load_async_to_lds_b128(
        (__attribute__((address_space(1))) vint4*)g,
        (__attribute__((address_space(3))) vint4*)l, 0, 0);
#else
    unsigned lds_off = (unsigned)(unsigned long long)(void*)l;
    asm volatile("global_load_async_to_lds_b128 %0, %1, off"
                 :: "v"(lds_off), "v"((const void*)g) : "memory");
#endif
}

__device__ __forceinline__ void wait_async_all() {
    asm volatile("s_wait_asynccnt 0" ::: "memory");
}

// ---------------------------------------------------------------------------
// Prep: fp32 -> scaled f16, row-major copy + transposed copy
// ---------------------------------------------------------------------------
__global__ __launch_bounds__(256) void prep_kernel(const float* __restrict__ q,
                                                   _Float16* __restrict__ qh,
                                                   _Float16* __restrict__ qhT,
                                                   int L, size_t total) {
    size_t idx = (size_t)blockIdx.x * blockDim.x + threadIdx.x;
    if (idx >= total) return;
    int d      = (int)(idx % HD);
    size_t bl  = idx / HD;
    int l      = (int)(bl % (size_t)L);
    int b      = (int)(bl / (size_t)L);
    _Float16 h = (_Float16)(q[idx] * SCALE);
    qh[idx] = h;
    qhT[((size_t)b * HD + d) * (size_t)L + l] = h;
}

// ---------------------------------------------------------------------------
// One flash pass: Out[l,:] = softmax_m( Qh[l]·Kh[m] + NEG_BIG*mq[l]*mk[m] ) @ Kh
// ---------------------------------------------------------------------------
__global__ __launch_bounds__(WAVES * 32) void
flash_pass(const _Float16* __restrict__ Qh,
           const _Float16* __restrict__ Kh,
           const _Float16* __restrict__ KhT,
           const int* __restrict__ mq,
           const int* __restrict__ mk,
           float* __restrict__ Out,
           int L, int M) {
    __shared__ _Float16 qs[WAVES][16 * QS_STRIDE];   // per-wave Q tile   (33.0 KB)
    __shared__ _Float16 ps[WAVES][16 * 32];          // per-wave P stage  ( 4.0 KB)
    __shared__ _Float16 kbuf[2][32 * KS_STRIDE];     // shared K tiles    (33.0 KB)
    __shared__ _Float16 ktbuf[2][HD * KT_STRIDE];    // shared K^T tiles  (40.0 KB)

    const int tid   = threadIdx.x;
    const int wave  = tid >> 5;
    const int lane  = tid & 31;
    const int hi    = lane >> 4;
    const int ln    = lane & 15;
    const int b     = blockIdx.y;
    const int qrow0 = (blockIdx.x * WAVES + wave) * 16;

    Qh  += (size_t)b * L * HD;
    Kh  += (size_t)b * M * HD;
    KhT += (size_t)b * HD * M;
    mq  += (size_t)b * L;
    mk  += (size_t)b * M;
    Out += (size_t)b * L * HD;

    // ---- prologue: async-copy my wave's Q tile (16 x 256 halfs, padded rows)
    {
        const _Float16* src = Qh + (size_t)qrow0 * HD;
        #pragma unroll
        for (int i = 0; i < 16; ++i)                 // row i, 16B chunk = lane
            async_cp16(src + i * HD + lane * 8,
                       &qs[wave][i * QS_STRIDE + lane * 8]);
    }

    // ---- async tile fetch: K tile (32 x 256) and K^T tile (256 x 32) -------
    auto issue_tiles = [&](int m0n, int nb) {
        const _Float16* ksrc = Kh + (size_t)m0n * HD;
        #pragma unroll
        for (int i = 0; i < 8; ++i) {                // 1024 chunks / 128 thr
            int cid = tid + 128 * i;
            int row = cid >> 5, c = cid & 31;
            async_cp16(ksrc + row * HD + c * 8,
                       &kbuf[nb][row * KS_STRIDE + c * 8]);
        }
        const _Float16* ktsrc = KhT + m0n;
        #pragma unroll
        for (int i = 0; i < 8; ++i) {                // 1024 chunks / 128 thr
            int cid = tid + 128 * i;
            int d = cid >> 2, c = cid & 3;
            async_cp16(ktsrc + (size_t)d * M + c * 8,
                       &ktbuf[nb][d * KT_STRIDE + c * 8]);
        }
    };
    issue_tiles(0, 0);

    // ---- per-row state (overlaps with in-flight async loads) ----
    float pm[8];
    #pragma unroll
    for (int r = 0; r < 8; ++r)
        pm[r] = NEG_BIG * (float)mq[qrow0 + r + 8 * hi];

    float mi[8], li[8];
    #pragma unroll
    for (int r = 0; r < 8; ++r) { mi[r] = -3.0e38f; li[r] = 0.0f; }

    v8f vzero = {};
    v8f acc[16];
    #pragma unroll
    for (int j = 0; j < 16; ++j) acc[j] = vzero;

    const int NIT = M / 32;
    for (int it = 0; it < NIT; ++it) {
        const int cur = it & 1;
        const int m0  = it * 32;

        wait_async_all();        // my slice of buf[cur] (and Q on it==0) landed
        __syncthreads();         // everyone's slice landed; buf[cur^1] free
        if (it + 1 < NIT) issue_tiles(m0 + 32, cur ^ 1);

        // ---------- S tile (16 x 32) = Q(16x256) * K^T(256x32), from LDS ----
        v8f s0 = vzero, s1 = vzero;
        const _Float16* kr0 = &kbuf[cur][(size_t)ln        * KS_STRIDE + hi * 8];
        const _Float16* kr1 = &kbuf[cur][(size_t)(16 + ln) * KS_STRIDE + hi * 8];
        #pragma unroll
        for (int kk = 0; kk < 8; ++kk) {
            const int kb = kk * 32;
            v8h alo = *(const v8h*)&qs[wave][ln * QS_STRIDE + kb + hi * 8];
            v8h ahi = *(const v8h*)&qs[wave][ln * QS_STRIDE + kb + 16 + hi * 8];
            v16h a  = __builtin_shufflevector(alo, ahi,
                        0,1,2,3,4,5,6,7,8,9,10,11,12,13,14,15);
            v8h b0l = *(const v8h*)(kr0 + kb);
            v8h b0h = *(const v8h*)(kr0 + kb + 16);
            v8h b1l = *(const v8h*)(kr1 + kb);
            v8h b1h = *(const v8h*)(kr1 + kb + 16);
            v16h b0 = __builtin_shufflevector(b0l, b0h,
                        0,1,2,3,4,5,6,7,8,9,10,11,12,13,14,15);
            v16h b1 = __builtin_shufflevector(b1l, b1h,
                        0,1,2,3,4,5,6,7,8,9,10,11,12,13,14,15);
            s0 = __builtin_amdgcn_wmma_f32_16x16x32_f16(false, a, false, b0,
                                                        (short)0, s0, false, false);
            s1 = __builtin_amdgcn_wmma_f32_16x16x32_f16(false, a, false, b1,
                                                        (short)0, s1, false, false);
        }

        // ---------- mask ----------
        const float ck0 = (float)mk[m0 + ln];
        const float ck1 = (float)mk[m0 + 16 + ln];
        #pragma unroll
        for (int r = 0; r < 8; ++r) {
            s0[r] += pm[r] * ck0;
            s1[r] += pm[r] * ck1;
        }

        // ---------- online softmax (row = 16 lanes of one half-wave) --------
        float al[8];
        #pragma unroll
        for (int r = 0; r < 8; ++r) {
            float t = fmaxf(s0[r], s1[r]);
            t = fmaxf(t, __shfl_xor(t, 1, 32));
            t = fmaxf(t, __shfl_xor(t, 2, 32));
            t = fmaxf(t, __shfl_xor(t, 4, 32));
            t = fmaxf(t, __shfl_xor(t, 8, 32));
            const float mnew = fmaxf(mi[r], t);
            al[r] = __expf(mi[r] - mnew);
            s0[r] = __expf(s0[r] - mnew);
            s1[r] = __expf(s1[r] - mnew);
            float rs = s0[r] + s1[r];
            rs += __shfl_xor(rs, 1, 32);
            rs += __shfl_xor(rs, 2, 32);
            rs += __shfl_xor(rs, 4, 32);
            rs += __shfl_xor(rs, 8, 32);
            li[r] = li[r] * al[r] + rs;
            mi[r] = mnew;
        }

        #pragma unroll
        for (int j = 0; j < 16; ++j)
            #pragma unroll
            for (int r = 0; r < 8; ++r)
                acc[j][r] *= al[r];

        // ---------- P: C layout -> LDS row-major (16x32 f16) -> A fragment --
        #pragma unroll
        for (int r = 0; r < 8; ++r) {
            const int m = r + 8 * hi;
            ps[wave][m * 32 + ln]      = (_Float16)s0[r];
            ps[wave][m * 32 + 16 + ln] = (_Float16)s1[r];
        }
        asm volatile("s_wait_dscnt 0" ::: "memory");

        v8h plo = *(const v8h*)&ps[wave][ln * 32 + hi * 8];
        v8h phi = *(const v8h*)&ps[wave][ln * 32 + 16 + hi * 8];
        v16h pa = __builtin_shufflevector(plo, phi,
                    0,1,2,3,4,5,6,7,8,9,10,11,12,13,14,15);

        // ---------- O += P(16x32) * V(32x256), B fragments from K^T tile ----
        #pragma unroll
        for (int j = 0; j < 16; ++j) {
            const _Float16* vb = &ktbuf[cur][(size_t)(j * 16 + ln) * KT_STRIDE + hi * 8];
            v8h bvl = *(const v8h*)(vb);
            v8h bvh = *(const v8h*)(vb + 16);
            v16h bv = __builtin_shufflevector(bvl, bvh,
                        0,1,2,3,4,5,6,7,8,9,10,11,12,13,14,15);
            acc[j] = __builtin_amdgcn_wmma_f32_16x16x32_f16(false, pa, false, bv,
                                                            (short)0, acc[j],
                                                            false, false);
        }
    }

    // ---------- epilogue: divide by row sums, store fp32 ----------
    float invl[8];
    #pragma unroll
    for (int r = 0; r < 8; ++r) invl[r] = 1.0f / li[r];

    #pragma unroll
    for (int j = 0; j < 16; ++j)
        #pragma unroll
        for (int r = 0; r < 8; ++r) {
            const int row = qrow0 + r + 8 * hi;
            Out[(size_t)row * HD + j * 16 + ln] = acc[j][r] * invl[r];
        }
}

// ---------------------------------------------------------------------------
extern "C" void kernel_launch(void* const* d_in, const int* in_sizes, int n_in,
                              void* d_out, int out_size, void* d_ws, size_t ws_size,
                              hipStream_t stream) {
    (void)in_sizes; (void)n_in; (void)out_size; (void)ws_size;
    const int B = 8, L = 2048, M = 2048;
    const size_t NE = (size_t)B * L * HD;

    const float* q0    = (const float*)d_in[0];
    const float* q1    = (const float*)d_in[1];
    const int*   mask0 = (const int*)d_in[4];
    const int*   mask1 = (const int*)d_in[5];
    float*       out   = (float*)d_out;

    _Float16* ws   = (_Float16*)d_ws;        // 32MB: q0h | q1h | q0hT | q1hT
    _Float16* q0h  = ws;
    _Float16* q1h  = ws + NE;
    _Float16* q0hT = ws + 2 * NE;
    _Float16* q1hT = ws + 3 * NE;

    const int pthreads = 256;
    const int pblocks  = (int)((NE + pthreads - 1) / pthreads);
    prep_kernel<<<pblocks, pthreads, 0, stream>>>(q0, q0h, q0hT, L, NE);
    prep_kernel<<<pblocks, pthreads, 0, stream>>>(q1, q1h, q1hT, M, NE);

    dim3 grid(L / (WAVES * 16), B);
    flash_pass<<<grid, WAVES * 32, 0, stream>>>(q0h, q1h, q1hT, mask0, mask1,
                                                out, L, M);
    flash_pass<<<grid, WAVES * 32, 0, stream>>>(q1h, q0h, q0hT, mask1, mask0,
                                                out + NE, M, L);
}
